// MambaBlock_34102040330337
// MI455X (gfx1250) — matrix operand order repaired and checked by reading, not compiled, not run on hardware
//
#include <hip/hip_runtime.h>

// ---------------------------------------------------------------------------
// Mamba block for MI455X (gfx1250), wave32, WMMA bf16 16x16x32 path.
// Each wave computes a 16(M) x 16*NW(N) tile: one A fragment is reused by NW
// B fragments -> NW back-to-back v_wmma per K-step (4x matrix-op density).
// ---------------------------------------------------------------------------

typedef __attribute__((ext_vector_type(16))) __bf16 v16bf;
typedef __attribute__((ext_vector_type(8)))  __bf16 v8bf;
typedef __attribute__((ext_vector_type(8)))  float  v8f;

#define B_   2
#define L_   1024
#define DM_  1024
#define DI_  2048
#define NS_  16
#define DTR_ 64
#define TOK_ (B_ * L_)      // 2048 tokens

__device__ __forceinline__ __bf16 f2bf(float f) {
    unsigned int u = __float_as_uint(f);
    unsigned int r = (u + 0x7FFFu + ((u >> 16) & 1u)) >> 16;   // RNE
    unsigned short s = (unsigned short)r;
    return __builtin_bit_cast(__bf16, s);
}

__device__ __forceinline__ float silu_f(float x) {
    return x / (1.0f + __expf(-x));
}

// ---------------------------------------------------------------------------
// fp32 -> bf16 conversion (grid-stride)
// ---------------------------------------------------------------------------
__global__ void cvt_bf16_kernel(const float* __restrict__ src,
                                __bf16* __restrict__ dst, int n) {
    for (int i = blockIdx.x * blockDim.x + threadIdx.x; i < n;
         i += gridDim.x * blockDim.x)
        dst[i] = f2bf(src[i]);
}

// bias96 = [dtin_b (64) | zeros (32)]
__global__ void bias96_kernel(const float* __restrict__ dtin_b,
                              float* __restrict__ bias96) {
    int i = threadIdx.x;
    if (i < 96) bias96[i] = (i < DTR_) ? dtin_b[i] : 0.0f;
}

// Aneg = -exp(A_log), (DI, NS)
__global__ void aneg_kernel(const float* __restrict__ a_log,
                            float* __restrict__ aneg, int n) {
    for (int i = blockIdx.x * blockDim.x + threadIdx.x; i < n;
         i += gridDim.x * blockDim.x)
        aneg[i] = -__expf(a_log[i]);
}

// ---------------------------------------------------------------------------
// LayerNorm -> bf16 activations. One block (256 thr) per token.
// ---------------------------------------------------------------------------
__global__ void ln_kernel(const float* __restrict__ x,
                          const float* __restrict__ g,
                          const float* __restrict__ b,
                          __bf16* __restrict__ xn) {
    const int t   = blockIdx.x;
    const int tid = threadIdx.x;
    const float* row = x + (size_t)t * DM_;

    float s = 0.f, s2 = 0.f;
    for (int j = tid; j < DM_; j += 256) { float v = row[j]; s += v; s2 += v * v; }
#pragma unroll
    for (int o = 16; o > 0; o >>= 1) {
        s  += __shfl_xor(s,  o, 32);
        s2 += __shfl_xor(s2, o, 32);
    }
    __shared__ float sh[16];
    const int wv = tid >> 5;
    if ((tid & 31) == 0) { sh[wv] = s; sh[wv + 8] = s2; }
    __syncthreads();
    if (tid == 0) {
        float a = 0.f, c = 0.f;
        for (int i = 0; i < 8; ++i) { a += sh[i]; c += sh[i + 8]; }
        float mu  = a * (1.0f / DM_);
        float var = c * (1.0f / DM_) - mu * mu;
        sh[0] = mu;
        sh[1] = rsqrtf(var + 1e-5f);
    }
    __syncthreads();
    const float mu = sh[0], rs = sh[1];
    __bf16* orow = xn + (size_t)t * DM_;
    for (int j = tid; j < DM_; j += 256)
        orow[j] = f2bf((row[j] - mu) * rs * g[j] + b[j]);
}

// ---------------------------------------------------------------------------
// WMMA bf16 GEMM:  C[M,N] = A[M,K] @ W[N,K]^T  (+bias[n]) (+softplus) (+res)
// Wave tile: 16 x (16*NW).  One A fragment reused across NW accumulators ->
// NW consecutive v_wmma_f32_16x16x32_bf16 per K-step.
// blockDim = 256 (8 waves).  act: 0 = none, 1 = softplus.
// ---------------------------------------------------------------------------
template <int NW>
__global__ void gemm_wmma_bf16(const __bf16* __restrict__ A,
                               const __bf16* __restrict__ W,
                               const float* __restrict__ bias,
                               const float* __restrict__ res,
                               float* __restrict__ C,
                               int M, int N, int K, int act) {
    const int lane   = threadIdx.x & 31;
    const int wave   = threadIdx.x >> 5;
    const int groupsN = N / (16 * NW);
    const int group  = blockIdx.x * (blockDim.x >> 5) + wave;
    const int total  = (M >> 4) * groupsN;
    if (group >= total) return;              // uniform per wave: EXEC stays all-1

    const int m0 = (group / groupsN) << 4;
    const int n0 = (group % groupsN) * (16 * NW);

    // A fragment: row = m0 + (lane&15); halves at K-offsets aSel and 16+aSel
    const int aSel = (lane >> 4) << 3;       // 0 or 8
    // B fragment: row of W = n tile + (lane&15); halves at K-offsets bSel and 16+bSel
    const int bSel = (lane >> 4) << 3;       // 0 or 8

    const __bf16* pa = A + (size_t)(m0 + (lane & 15)) * K + aSel;
    const __bf16* pb[NW];
#pragma unroll
    for (int j = 0; j < NW; ++j)
        pb[j] = W + (size_t)(n0 + j * 16 + (lane & 15)) * K + bSel;

    v8f acc[NW];
#pragma unroll
    for (int j = 0; j < NW; ++j) acc[j] = (v8f){};

    for (int k = 0; k < K; k += 32) {
        const v8bf alo = *reinterpret_cast<const v8bf*>(pa + k);
        const v8bf ahi = *reinterpret_cast<const v8bf*>(pa + k + 16);
        v16bf afr;
#pragma unroll
        for (int i = 0; i < 8; ++i) { afr[i] = alo[i]; afr[i + 8] = ahi[i]; }

        v16bf bfr[NW];
#pragma unroll
        for (int j = 0; j < NW; ++j) {
            const v8bf blo = *reinterpret_cast<const v8bf*>(pb[j] + k);
            const v8bf bhi = *reinterpret_cast<const v8bf*>(pb[j] + k + 16);
#pragma unroll
            for (int i = 0; i < 8; ++i) { bfr[j][i] = blo[i]; bfr[j][i + 8] = bhi[i]; }
        }

        if (k + 32 < K) {                    // stream next K tile
            __builtin_prefetch(pa + k + 64, 0, 1);   // -> global_prefetch_b8
#pragma unroll
            for (int j = 0; j < NW; ++j)
                __builtin_prefetch(pb[j] + k + 64, 0, 1);
        }

#pragma unroll
        for (int j = 0; j < NW; ++j)
            acc[j] = __builtin_amdgcn_wmma_f32_16x16x32_bf16(
                         false, afr, false, bfr[j], (short)0, acc[j],
                         false, false);
    }

    const int mC = m0 + ((lane >> 4) << 3);
#pragma unroll
    for (int j = 0; j < NW; ++j) {
        const int nC = n0 + j * 16 + (lane & 15);
        const float bv = bias ? bias[nC] : 0.0f;
#pragma unroll
        for (int i = 0; i < 8; ++i) {
            float v = acc[j][i] + bv;
            if (act == 1) v = (v > 20.0f) ? v : log1pf(__expf(v));
            const size_t idx = (size_t)(mC + i) * N + nC;
            if (res) v += res[idx];
            C[idx] = v;
        }
    }
}

// ---------------------------------------------------------------------------
// Causal depthwise conv1d (4-tap) + SiLU.  xz row stride 2*DI; x_proj = cols
// [0, DI).  Writes fp32 (scan input u) and bf16 (GEMM A operand).
// ---------------------------------------------------------------------------
__global__ void conv_silu_kernel(const float* __restrict__ xz,
                                 const float* __restrict__ conv_w,
                                 const float* __restrict__ conv_b,
                                 float* __restrict__ xc,
                                 __bf16* __restrict__ xc_bf) {
    for (int idx = blockIdx.x * blockDim.x + threadIdx.x; idx < TOK_ * DI_;
         idx += gridDim.x * blockDim.x) {
        const int c  = idx % DI_;
        const int t  = idx / DI_;
        const int l  = t % L_;
        const int bb = t / L_;
        float acc = conv_b[c];
#pragma unroll
        for (int k = 0; k < 4; ++k) {
            const int ls = l + k - 3;
            if (ls >= 0)
                acc += conv_w[c * 4 + k] *
                       xz[((size_t)(bb * L_ + ls)) * (2 * DI_) + c];
        }
        const float s = silu_f(acc);
        xc[idx]    = s;
        xc_bf[idx] = f2bf(s);
    }
}

// Unpack fused (2048 x 96) projection: cols [0,64) -> dt_in bf16,
// [64,80) -> Bm fp32, [80,96) -> Cm fp32.
__global__ void pack96_kernel(const float* __restrict__ p96,
                              __bf16* __restrict__ dtin_bf,
                              float* __restrict__ Bm,
                              float* __restrict__ Cm) {
    for (int idx = blockIdx.x * blockDim.x + threadIdx.x; idx < TOK_ * 96;
         idx += gridDim.x * blockDim.x) {
        const int t = idx / 96, c = idx % 96;
        const float v = p96[idx];
        if (c < 64)       dtin_bf[t * 64 + c]    = f2bf(v);
        else if (c < 80)  Bm[t * NS_ + (c - 64)] = v;
        else              Cm[t * NS_ + (c - 80)] = v;
    }
}

// ---------------------------------------------------------------------------
// Selective scan.  Lane = one (d, n) state element; 16-lane shuffle tree
// reduces over n.  Block = 256 thr = 16 d-channels; grid = B * DI/16.
// y includes the skip term u*D.
// ---------------------------------------------------------------------------
__global__ void scan_kernel(const float* __restrict__ u,
                            const float* __restrict__ dt,
                            const float* __restrict__ Bm,
                            const float* __restrict__ Cm,
                            const float* __restrict__ Aneg,
                            const float* __restrict__ Dp,
                            float* __restrict__ y) {
    const int tid  = threadIdx.x;
    const int n    = tid & 15;
    const int dloc = tid >> 4;                       // 0..15
    const int b    = blockIdx.x >> 7;                // 128 blocks per batch
    const int d    = ((blockIdx.x & 127) << 4) + dloc;

    const float Ad = Aneg[d * NS_ + n];
    const float Dv = Dp[d];
    float h = 0.0f;
    const size_t tb = (size_t)b * L_;
    for (int l = 0; l < L_; ++l) {
        const size_t t = tb + l;
        const float dtv = dt[t * DI_ + d];
        const float uv  = u [t * DI_ + d];
        const float Bv  = Bm[t * NS_ + n];
        const float Cv  = Cm[t * NS_ + n];
        h = h * __expf(dtv * Ad) + dtv * uv * Bv;
        float p = h * Cv;
        p += __shfl_xor(p, 1, 32);
        p += __shfl_xor(p, 2, 32);
        p += __shfl_xor(p, 4, 32);
        p += __shfl_xor(p, 8, 32);
        if (n == 0) y[t * DI_ + d] = p + uv * Dv;
    }
}

// Gate: yg = bf16( y * silu(z) ),  z = xz cols [DI, 2*DI)
__global__ void gate_kernel(const float* __restrict__ y,
                            const float* __restrict__ xz,
                            __bf16* __restrict__ yg) {
    for (int idx = blockIdx.x * blockDim.x + threadIdx.x; idx < TOK_ * DI_;
         idx += gridDim.x * blockDim.x) {
        const int t = idx / DI_, c = idx % DI_;
        const float z = xz[(size_t)t * (2 * DI_) + DI_ + c];
        yg[idx] = f2bf(y[idx] * silu_f(z));
    }
}

// ---------------------------------------------------------------------------
// Host side
// ---------------------------------------------------------------------------
extern "C" void kernel_launch(void* const* d_in, const int* in_sizes, int n_in,
                              void* d_out, int out_size, void* d_ws, size_t ws_size,
                              hipStream_t stream) {
    (void)in_sizes; (void)n_in; (void)out_size; (void)ws_size;

    const float* x       = (const float*)d_in[0];   // (2,1024,1024)
    const float* ln_g    = (const float*)d_in[1];
    const float* ln_b    = (const float*)d_in[2];
    const float* inW     = (const float*)d_in[3];   // (4096,1024)
    const float* conv_w  = (const float*)d_in[4];   // (2048,1,4)
    const float* conv_b  = (const float*)d_in[5];
    const float* dtin_w  = (const float*)d_in[6];   // (64,2048)
    const float* dtin_b  = (const float*)d_in[7];
    const float* dt_w    = (const float*)d_in[8];   // (2048,64)
    const float* dt_b    = (const float*)d_in[9];
    const float* B_w     = (const float*)d_in[10];  // (16,2048)
    const float* C_w     = (const float*)d_in[11];  // (16,2048)
    const float* A_log   = (const float*)d_in[12];  // (2048,16)
    const float* Dp      = (const float*)d_in[13];
    const float* outW    = (const float*)d_in[14];  // (1024,2048)
    float*       out     = (float*)d_out;

    // bump allocator over workspace
    size_t off = 0;
    char* base = (char*)d_ws;
    auto alloc = [&](size_t bytes) -> void* {
        off = (off + 255) & ~(size_t)255;
        void* p = base + off;
        off += bytes;
        return p;
    };

    __bf16* xn_bf    = (__bf16*)alloc((size_t)TOK_ * DM_ * 2);
    __bf16* wIn_bf   = (__bf16*)alloc((size_t)(2 * DI_) * DM_ * 2);
    __bf16* w96_bf   = (__bf16*)alloc((size_t)96 * DI_ * 2);
    __bf16* wDt_bf   = (__bf16*)alloc((size_t)DI_ * DTR_ * 2);
    __bf16* wOut_bf  = (__bf16*)alloc((size_t)DM_ * DI_ * 2);
    float*  bias96   = (float*) alloc(96 * 4);
    float*  xz       = (float*) alloc((size_t)TOK_ * 2 * DI_ * 4);
    float*  xconv    = (float*) alloc((size_t)TOK_ * DI_ * 4);
    __bf16* xconv_bf = (__bf16*)alloc((size_t)TOK_ * DI_ * 2);
    float*  proj96   = (float*) alloc((size_t)TOK_ * 96 * 4);
    __bf16* dtin_bf  = (__bf16*)alloc((size_t)TOK_ * DTR_ * 2);
    float*  Bm       = (float*) alloc((size_t)TOK_ * NS_ * 4);
    float*  Cm       = (float*) alloc((size_t)TOK_ * NS_ * 4);
    float*  dtbuf    = (float*) alloc((size_t)TOK_ * DI_ * 4);
    float*  Aneg     = (float*) alloc((size_t)DI_ * NS_ * 4);
    float*  yscan    = (float*) alloc((size_t)TOK_ * DI_ * 4);
    __bf16* yg_bf    = (__bf16*)alloc((size_t)TOK_ * DI_ * 2);

    const int T = 256;
    const int CG = 2048;   // grid for grid-stride kernels

    // weight conversions fp32 -> bf16 (w96 packs dtin_w | B_w | C_w, K-major)
    cvt_bf16_kernel<<<CG, T, 0, stream>>>(inW,    wIn_bf,  2 * DI_ * DM_);
    cvt_bf16_kernel<<<CG, T, 0, stream>>>(dtin_w, w96_bf,            DTR_ * DI_);
    cvt_bf16_kernel<<<CG, T, 0, stream>>>(B_w,    w96_bf + (size_t)64 * DI_, NS_ * DI_);
    cvt_bf16_kernel<<<CG, T, 0, stream>>>(C_w,    w96_bf + (size_t)80 * DI_, NS_ * DI_);
    cvt_bf16_kernel<<<CG, T, 0, stream>>>(dt_w,   wDt_bf,  DI_ * DTR_);
    cvt_bf16_kernel<<<CG, T, 0, stream>>>(outW,   wOut_bf, DM_ * DI_);
    bias96_kernel<<<1, 96, 0, stream>>>(dtin_b, bias96);
    aneg_kernel<<<64, T, 0, stream>>>(A_log, Aneg, DI_ * NS_);

    // 1) LayerNorm -> bf16
    ln_kernel<<<TOK_, T, 0, stream>>>(x, ln_g, ln_b, xn_bf);

    // 2) GEMM1: xz = xn @ inW^T   (M=2048, N=4096, K=1024), 16x64 wave tiles
    {
        int groups = (TOK_ / 16) * ((2 * DI_) / 64);
        gemm_wmma_bf16<4><<<(groups + 7) / 8, T, 0, stream>>>(
            xn_bf, wIn_bf, nullptr, nullptr, xz, TOK_, 2 * DI_, DM_, 0);
    }

    // 3) causal depthwise conv + SiLU
    conv_silu_kernel<<<CG * 4, T, 0, stream>>>(xz, conv_w, conv_b, xconv, xconv_bf);

    // 4) GEMM2 (fused dt_in | B | C): (M=2048, N=96, K=2048), 16x32 wave tiles
    {
        int groups = (TOK_ / 16) * (96 / 32);
        gemm_wmma_bf16<2><<<(groups + 7) / 8, T, 0, stream>>>(
            xconv_bf, w96_bf, bias96, nullptr, proj96, TOK_, 96, DI_, 0);
    }
    pack96_kernel<<<CG, T, 0, stream>>>(proj96, dtin_bf, Bm, Cm);

    // 5) GEMM3: dt = softplus(dt_in @ dt_w^T + dt_b)  (M=2048, N=2048, K=64)
    {
        int groups = (TOK_ / 16) * (DI_ / 64);
        gemm_wmma_bf16<4><<<(groups + 7) / 8, T, 0, stream>>>(
            dtin_bf, wDt_bf, dt_b, nullptr, dtbuf, TOK_, DI_, DTR_, 1);
    }

    // 6) selective scan (includes + u*D)
    scan_kernel<<<B_ * (DI_ / 16), T, 0, stream>>>(
        xconv, dtbuf, Bm, Cm, Aneg, Dp, yscan);

    // 7) gate: yg = y * silu(z)
    gate_kernel<<<CG * 4, T, 0, stream>>>(yscan, xz, yg_bf);

    // 8) GEMM4: out = yg @ outW^T + residual  (M=2048, N=1024, K=2048)
    {
        int groups = (TOK_ / 16) * (DM_ / 64);
        gemm_wmma_bf16<4><<<(groups + 7) / 8, T, 0, stream>>>(
            yg_bf, wOut_bf, nullptr, x, out, TOK_, DM_, DI_, 0);
    }
}
